// EpisodeMultiheadAttentionBlock_89026082111911
// MI455X (gfx1250) — matrix operand order repaired and checked by reading, not compiled, not run on hardware
//
#include <hip/hip_runtime.h>
#include <hip/hip_bf16.h>
#include <math.h>

typedef __attribute__((ext_vector_type(16))) _Float16 v16h;
typedef __attribute__((ext_vector_type(8)))  _Float16 v8h;
typedef __attribute__((ext_vector_type(8)))  float    v8f;

#define B_  16
#define L_  1024
#define E_  1024
#define H_  16
#define HD_ 64
#define QL_ 256
#define R_  768   /* L - QL */

#define WMMA_F16(a,b,c) __builtin_amdgcn_wmma_f32_16x16x32_f16(false,(a),false,(b),(short)0,(c),false,false)

// A-fragment (16x32, MxK): lane<16 holds K {0..7,16..23}; lane>=16 holds K {8..15,24..31}
__device__ __forceinline__ v16h ld_fragA(const _Float16* row, int hs) {
  const _Float16* p = row + hs * 8;
  v8h lo = *(const v8h*)(p);
  v8h hi = *(const v8h*)(p + 16);
  v16h f;
#pragma unroll
  for (int i = 0; i < 8; ++i) { f[i] = lo[i]; f[i + 8] = hi[i]; }
  return f;
}

// B-fragment (32x16, KxN) from W rows (X@W^T): lane<16 holds K {0..7,16..23}; lane>=16 holds K {8..15,24..31}
__device__ __forceinline__ v16h ld_fragB(const _Float16* row, int hs) {
  const _Float16* p = row + hs * 8;
  v8h lo = *(const v8h*)(p);
  v8h hi = *(const v8h*)(p + 16);
  v16h f;
#pragma unroll
  for (int i = 0; i < 8; ++i) { f[i] = lo[i]; f[i + 8] = hi[i]; }
  return f;
}

// ---- CDNA5 async global->LDS copy (16B per lane), tracked by ASYNCcnt ----
// LDS offset = low 32 bits of the generic pointer (aperture scheme: addr[31:0] is the LDS address)
__device__ __forceinline__ unsigned lds_off(const void* p) {
  return (unsigned)(unsigned long long)p;
}
__device__ __forceinline__ void async_cp16(unsigned lds_byte_off, const _Float16* g) {
  asm volatile("global_load_async_to_lds_b128 %0, %1, off"
               :: "v"(lds_byte_off), "v"((unsigned long long)g) : "memory");
}
__device__ __forceinline__ void async_wait0() {
  asm volatile("s_wait_asynccnt 0" ::: "memory");
}

// ---------------- small utility kernels ----------------
__global__ void cvt_f32_f16(const float* __restrict__ src, _Float16* __restrict__ dst, int n) {
  int i = blockIdx.x * blockDim.x + threadIdx.x;
  if (i < n) dst[i] = (_Float16)src[i];
}

__global__ void pack_x(const float* __restrict__ key, const float* __restrict__ pe,
                       _Float16* __restrict__ xk, _Float16* __restrict__ xv, int n) {
  int i = blockIdx.x * blockDim.x + threadIdx.x;
  if (i < n) {
    float k = key[i];
    xk[i] = (_Float16)(k + pe[i]);
    xv[i] = (_Float16)k;
  }
}

__global__ void zero_f32(float* __restrict__ p, int n) {
  int i = blockIdx.x * blockDim.x + threadIdx.x;
  if (i < n) p[i] = 0.f;
}

// ---------------- LDS double-buffered WMMA GEMM: D = act(A @ W^T + bias [+Cin]) [+res] ----------------
// block = 256 threads (8 waves). Block tile 128(m) x 128(n), K-slab 32, double buffered in LDS.
// Per K-slab each thread issues 4x global_load_async_to_lds_b128 for the NEXT slab while the
// current slab feeds 8 WMMAs per wave from ds_load_b128.
// aQMap/addQMap: row m -> (m/QL)*L + 768 + (m%QL). vtMode: store D16 as (B,H,64,L).
// Gr!=null => fused gating epilogue: D32 = (1-z)*oriq + z*tanh(r*U + val + bias2)
__global__ __launch_bounds__(256) void gemm_xwt(
    const _Float16* __restrict__ A, int lda, int aQMap,
    const _Float16* __restrict__ W, int ldw,
    const float* __restrict__ bias,
    const float* __restrict__ Cin,
    float* __restrict__ D32,
    _Float16* __restrict__ D16,
    int M, int N, int K,
    int relu, float scale,
    const float* __restrict__ addRes, int addQMap,
    int vtMode,
    const float* __restrict__ Gr, const float* __restrict__ Gz,
    const float* __restrict__ Uu, const float* __restrict__ bias2)
{
  __shared__ __align__(16) _Float16 sA[2][128 * 32];
  __shared__ __align__(16) _Float16 sB[2][128 * 32];

  const int tid  = threadIdx.x;
  const int lane = tid & 31;
  const int wv   = tid >> 5;
  const int r16  = lane & 15;
  const int hs   = lane >> 4;
  const int wm   = wv & 3;          // 4 groups of 32 rows
  const int wn   = wv >> 2;         // 2 groups of 64 cols
  const int m0   = blockIdx.y * 128;
  const int n0   = blockIdx.x * 128;

  // staging: thread covers halfs [(tid&1)*16, +16) of tile row (tid>>1)
  const int trow = tid >> 1;
  const int toff = (tid & 1) * 16;
  const int am   = m0 + trow;
  const int arow = aQMap ? (((am >> 8) << 10) + R_ + (am & (QL_ - 1))) : am;
  const _Float16* gA = A + (size_t)arow * lda + toff;
  const _Float16* gB = W + (size_t)(n0 + trow) * ldw + toff;
  const unsigned stOff = (unsigned)(trow * 32 + toff) * 2u;  // byte offset inside one buffer
  const unsigned aBase[2] = { lds_off(&sA[0][0]), lds_off(&sA[1][0]) };
  const unsigned bBase[2] = { lds_off(&sB[0][0]), lds_off(&sB[1][0]) };

  const int nsteps = K >> 5;

  // preload slab 0 into buffer 0
  async_cp16(aBase[0] + stOff,      gA);
  async_cp16(aBase[0] + stOff + 16, gA + 8);
  async_cp16(bBase[0] + stOff,      gB);
  async_cp16(bBase[0] + stOff + 16, gB + 8);
  async_wait0();
  __syncthreads();

  const v8f z8 = {0.f,0.f,0.f,0.f,0.f,0.f,0.f,0.f};
  v8f acc[2][4] = {{z8, z8, z8, z8}, {z8, z8, z8, z8}};

  for (int kt = 0; kt < nsteps; ++kt) {
    const int cur = kt & 1;
    if (kt + 1 < nsteps) {
      const int nb = cur ^ 1;
      const _Float16* gA2 = gA + (size_t)(kt + 1) * 32;
      const _Float16* gB2 = gB + (size_t)(kt + 1) * 32;
      async_cp16(aBase[nb] + stOff,      gA2);
      async_cp16(aBase[nb] + stOff + 16, gA2 + 8);
      async_cp16(bBase[nb] + stOff,      gB2);
      async_cp16(bBase[nb] + stOff + 16, gB2 + 8);
    }
    const _Float16* sa = sA[cur];
    const _Float16* sb = sB[cur];
    v16h bf[4];
#pragma unroll
    for (int t = 0; t < 4; ++t)
      bf[t] = ld_fragB(sb + (wn * 64 + t * 16 + r16) * 32, hs);
#pragma unroll
    for (int i = 0; i < 2; ++i) {
      v16h a = ld_fragA(sa + (wm * 32 + i * 16 + r16) * 32, hs);
#pragma unroll
      for (int t = 0; t < 4; ++t)
        acc[i][t] = WMMA_F16(a, bf[t], acc[i][t]);
    }
    async_wait0();
    __syncthreads();
  }

#pragma unroll
  for (int i = 0; i < 2; ++i) {
#pragma unroll
    for (int t = 0; t < 4; ++t) {
#pragma unroll
      for (int g = 0; g < 8; ++g) {
        const int m = m0 + wm * 32 + i * 16 + g + 8 * hs;
        const int n = n0 + wn * 64 + t * 16 + r16;
        const size_t idx = (size_t)m * N + n;
        float val = acc[i][t][g];
        if (Cin)  val += Cin[idx];
        if (bias) val += bias[n];
        if (relu) val = fmaxf(val, 0.f);
        if (Gr) {
          const float rv = Gr[idx], zv = Gz[idx];
          const int rr = ((m >> 8) << 10) + R_ + (m & (QL_ - 1));
          const float oq = addRes[(size_t)rr * E_ + n];
          const float hh = tanhf(rv * Uu[idx] + val + bias2[n]);
          D32[idx] = (1.f - zv) * oq + zv * hh;
        } else {
          if (addRes) {
            const int rr = addQMap ? (((m >> 8) << 10) + R_ + (m & (QL_ - 1))) : m;
            val += addRes[(size_t)rr * E_ + n];
          }
          val *= scale;
          if (D32) D32[idx] = val;
          if (D16) {
            if (vtMode) {
              const int bb = m >> 10, l = m & (L_ - 1);
              const int hh2 = n >> 6, d = n & (HD_ - 1);
              D16[(size_t)((bb * H_ + hh2) * HD_ + d) * L_ + l] = (_Float16)val;
            } else {
              D16[idx] = (_Float16)val;
            }
          }
        }
      }
    }
  }
}

// ---------------- fused masked attention: scores -> softmax -> attn_w -> attn@V ----------------
// grid = (B, QL/16), block = 256 (8 waves). Loops over the 16 heads.
__global__ __launch_bounds__(256) void attn_fused(
    const _Float16* __restrict__ Q16,   // (B*QL, E), pre-scaled by 1/8
    const _Float16* __restrict__ K16,   // (B*L, E)
    const _Float16* __restrict__ Vt16,  // (B, H, 64, L)
    const int* __restrict__ kidx,       // (B, L)
    const unsigned char* __restrict__ pad, // (B, L) bool
    _Float16* __restrict__ ctx16,       // (B*QL, E)
    float* __restrict__ aw)             // (B*QL, L), pre-zeroed
{
  extern __shared__ float smem[];
  float* sc   = smem;                 // 16*1024
  float* pv   = smem + 16 * 1024;     // 8*16*64
  float* red  = pv + 8 * 16 * 64;     // 16*16
  float* rowv = red + 256;            // 16

  const int tid  = threadIdx.x;
  const int lane = tid & 31, wv = tid >> 5;
  const int r16  = lane & 15, hs = lane >> 4;
  const int b    = blockIdx.x;
  const int q0   = blockIdx.y * 16;
  const int srow = tid >> 4, scol = tid & 15;

  const int* kib = kidx + b * L_;
  const unsigned char* pb = pad + b * L_;
  const v8f z8 = {0.f,0.f,0.f,0.f,0.f,0.f,0.f,0.f};

  for (int h = 0; h < H_; ++h) {
    // ---- 1) scores: wave wv covers key columns [wv*128, wv*128+128) ----
    {
      const _Float16* aRow  = Q16 + (size_t)(b * QL_ + q0 + r16) * E_ + h * HD_;
      const _Float16* kBase = K16 + (size_t)b * L_ * E_ + h * HD_;
      v8f acc[8] = {z8, z8, z8, z8, z8, z8, z8, z8};
      for (int k = 0; k < HD_; k += 32) {
        v16h a = ld_fragA(aRow + k, hs);
#pragma unroll
        for (int t = 0; t < 8; ++t) {
          const int n = wv * 128 + t * 16 + r16;
          v16h bfr = ld_fragB(kBase + (size_t)n * E_ + k, hs);
          acc[t] = WMMA_F16(a, bfr, acc[t]);
        }
      }
#pragma unroll
      for (int t = 0; t < 8; ++t) {
#pragma unroll
        for (int g = 0; g < 8; ++g) {
          const int m  = g + 8 * hs;
          const int j  = wv * 128 + t * 16 + r16;
          const int ig = R_ + q0 + m;   // global query row in [768,1024)
          const bool keep = (j == ig) ||
                            (j < R_ && (kib[ig] > kib[j]) && (pb[j] == 0));
          sc[m * L_ + j] = keep ? acc[t][g] : -1e30f;
        }
      }
    }
    __syncthreads();
    // ---- 2) softmax over 1024 keys per row + attn_w accumulation ----
    {
      float* row = sc + srow * L_;
      float mx = -3.4e38f;
      for (int s = 0; s < 64; ++s) mx = fmaxf(mx, row[scol + 16 * s]);
      red[srow * 16 + scol] = mx;
      __syncthreads();
      if (tid < 16) {
        float m2 = -3.4e38f;
        for (int c = 0; c < 16; ++c) m2 = fmaxf(m2, red[tid * 16 + c]);
        rowv[tid] = m2;
      }
      __syncthreads();
      const float rm = rowv[srow];
      float sum = 0.f;
      for (int s = 0; s < 64; ++s) {
        const int j = scol + 16 * s;
        const float e = __expf(row[j] - rm);
        row[j] = e;
        sum += e;
      }
      red[srow * 16 + scol] = sum;
      __syncthreads();
      if (tid < 16) {
        float s2 = 0.f;
        for (int c = 0; c < 16; ++c) s2 += red[tid * 16 + c];
        rowv[tid] = 1.f / s2;
      }
      __syncthreads();
      const float inv = rowv[srow];
      float* awr = aw + (size_t)(b * QL_ + q0 + srow) * L_;
      for (int s = 0; s < 64; ++s) {
        const int j = scol + 16 * s;
        const float p = row[j] * inv;
        row[j] = p;
        awr[j] += p * (1.0f / H_);   // mean over heads; same thread owns element every h
      }
    }
    __syncthreads();
    // ---- 3) PV partial: wave wv handles K range [wv*128, +128) ----
    {
      v8f acc[4] = {z8, z8, z8, z8};
      const _Float16* vBase = Vt16 + (size_t)((b * H_ + h) * HD_) * L_ + wv * 128;
      const float* arow = sc + r16 * L_ + wv * 128;
#pragma unroll
      for (int k = 0; k < 128; k += 32) {
        v16h a;
        const float* ap = arow + k + hs * 8;
#pragma unroll
        for (int i = 0; i < 8; ++i) { a[i] = (_Float16)ap[i]; a[i + 8] = (_Float16)ap[16 + i]; }
#pragma unroll
        for (int t = 0; t < 4; ++t) {
          v16h bfr = ld_fragB(vBase + (size_t)(t * 16 + r16) * L_ + k, hs);
          acc[t] = WMMA_F16(a, bfr, acc[t]);
        }
      }
#pragma unroll
      for (int t = 0; t < 4; ++t) {
#pragma unroll
        for (int g = 0; g < 8; ++g) {
          const int m = g + 8 * hs;
          const int n = t * 16 + r16;
          pv[(wv * 16 + m) * 64 + n] = acc[t][g];
        }
      }
    }
    __syncthreads();
    // ---- 4) reduce wave partials, store ctx slice for this head ----
    for (int e = tid; e < 16 * 64; e += 256) {
      float s = 0.f;
#pragma unroll
      for (int w2 = 0; w2 < 8; ++w2) s += pv[w2 * 1024 + e];
      const int m = e >> 6, n = e & 63;
      ctx16[(size_t)(b * QL_ + q0 + m) * E_ + h * HD_ + n] = (_Float16)s;
    }
    __syncthreads();
  }
}

// ---------------- host launcher ----------------
extern "C" void kernel_launch(void* const* d_in, const int* in_sizes, int n_in,
                              void* d_out, int out_size, void* d_ws, size_t ws_size,
                              hipStream_t stream) {
  (void)in_sizes; (void)n_in; (void)out_size; (void)ws_size;

  const float* key = (const float*)d_in[0];
  const float* pe  = (const float*)d_in[1];
  const int*   kidx = (const int*)d_in[2];
  const unsigned char* pad = (const unsigned char*)d_in[3];
  // d_in[4] = query_length (constant 256, compiled in)
  const float* in_proj_w = (const float*)d_in[5];
  const float* in_proj_b = (const float*)d_in[6];
  const float* out_w = (const float*)d_in[7];
  const float* out_b = (const float*)d_in[8];
  const float* wxr = (const float*)d_in[9];
  const float* bxr = (const float*)d_in[10];
  const float* wyr = (const float*)d_in[11];
  const float* byr = (const float*)d_in[12];
  const float* wxz = (const float*)d_in[13];
  const float* bxz = (const float*)d_in[14];
  const float* wyz = (const float*)d_in[15];
  const float* byz = (const float*)d_in[16];
  const float* wxg = (const float*)d_in[17];
  const float* bxg = (const float*)d_in[18];
  const float* wyg = (const float*)d_in[19];
  const float* byg = (const float*)d_in[20];

  const size_t E2 = (size_t)E_ * E_;
  _Float16* w16   = (_Float16*)d_ws;       // 10*E2 halfs
  _Float16* wq16  = w16;
  _Float16* wk16  = w16 + E2;
  _Float16* wv16  = w16 + 2 * E2;
  _Float16* wo16  = w16 + 3 * E2;
  _Float16* wxr16 = w16 + 4 * E2;
  _Float16* wyr16 = w16 + 5 * E2;
  _Float16* wxz16 = w16 + 6 * E2;
  _Float16* wyz16 = w16 + 7 * E2;
  _Float16* wxg16 = w16 + 8 * E2;
  _Float16* wyg16 = w16 + 9 * E2;
  _Float16* xk16  = w16 + 10 * E2;         // (B*L, E)
  _Float16* xv16  = xk16 + 16 * E2;        // (B*L, E)
  _Float16* q16   = xv16 + 16 * E2;        // (B*QL, E)
  _Float16* k16   = q16 + 4 * E2;          // (B*L, E)
  _Float16* vt16  = k16 + 16 * E2;         // (B, H, 64, L)
  _Float16* ctx16 = vt16 + 16 * E2;        // (B*QL, E)
  _Float16* out16 = ctx16 + 4 * E2;        // (B*QL, E)
  float* Gr = (float*)(out16 + 4 * E2);    // (B*QL, E) f32
  float* Gz = Gr + 4 * E2;
  float* Uu = Gz + 4 * E2;

  float* outp = (float*)d_out;                    // (B, QL, E)
  float* aw   = outp + (size_t)B_ * QL_ * E_;     // (B, QL, L)

  const int thr = 256;
  #define NB(n) (int)(((n) + thr - 1) / thr)

  // weights -> f16
  cvt_f32_f16<<<NB(3 * E2), thr, 0, stream>>>(in_proj_w, wq16, (int)(3 * E2));
  cvt_f32_f16<<<NB(E2), thr, 0, stream>>>(out_w, wo16, (int)E2);
  cvt_f32_f16<<<NB(E2), thr, 0, stream>>>(wxr, wxr16, (int)E2);
  cvt_f32_f16<<<NB(E2), thr, 0, stream>>>(wyr, wyr16, (int)E2);
  cvt_f32_f16<<<NB(E2), thr, 0, stream>>>(wxz, wxz16, (int)E2);
  cvt_f32_f16<<<NB(E2), thr, 0, stream>>>(wyz, wyz16, (int)E2);
  cvt_f32_f16<<<NB(E2), thr, 0, stream>>>(wxg, wxg16, (int)E2);
  cvt_f32_f16<<<NB(E2), thr, 0, stream>>>(wyg, wyg16, (int)E2);
  // activations: xk = f16(key+pe), xv = f16(key)
  pack_x<<<NB(16 * E2), thr, 0, stream>>>(key, pe, xk16, xv16, (int)(16 * E2));
  // zero attn_w output region (accumulated into)
  zero_f32<<<NB((size_t)B_ * QL_ * L_), thr, 0, stream>>>(aw, B_ * QL_ * L_);

  const dim3 blk(256);
  const int MQ = B_ * QL_, MK = B_ * L_;
  // Q = (q_in @ Wq^T + bq) * 1/8   (A = xk16 with ori-row remap)
  gemm_xwt<<<dim3(E_ / 128, MQ / 128), blk, 0, stream>>>(
      xk16, E_, 1, wq16, E_, in_proj_b, nullptr, nullptr, q16,
      MQ, E_, E_, 0, 0.125f, nullptr, 0, 0, nullptr, nullptr, nullptr, nullptr);
  // K = k_in @ Wk^T + bk
  gemm_xwt<<<dim3(E_ / 128, MK / 128), blk, 0, stream>>>(
      xk16, E_, 0, wk16, E_, in_proj_b + E_, nullptr, nullptr, k16,
      MK, E_, E_, 0, 1.f, nullptr, 0, 0, nullptr, nullptr, nullptr, nullptr);
  // V = v_in @ Wv^T + bv  -> stored transposed per head (B,H,64,L)
  gemm_xwt<<<dim3(E_ / 128, MK / 128), blk, 0, stream>>>(
      xv16, E_, 0, wv16, E_, in_proj_b + 2 * E_, nullptr, nullptr, vt16,
      MK, E_, E_, 0, 1.f, nullptr, 0, 1, nullptr, nullptr, nullptr, nullptr);
  // fused masked attention + softmax + attn_w + attn@V
  const size_t smem = (size_t)(16 * 1024 + 8 * 16 * 64 + 256 + 16) * sizeof(float);
  attn_fused<<<dim3(B_, QL_ / 16), dim3(256), smem, stream>>>(
      q16, k16, vt16, kidx, pad, ctx16, aw);
  // out = relu(ctx @ out_w^T + out_b) + ori_q
  gemm_xwt<<<dim3(E_ / 128, MQ / 128), blk, 0, stream>>>(
      ctx16, E_, 0, wo16, E_, out_b, nullptr, nullptr, out16,
      MQ, E_, E_, 1, 1.f, key, 1, 0, nullptr, nullptr, nullptr, nullptr);
  // r = relu(oriq@wxr^T + bxr + out@wyr^T + byr)  (two accumulating passes)
  gemm_xwt<<<dim3(E_ / 128, MQ / 128), blk, 0, stream>>>(
      xv16, E_, 1, wxr16, E_, bxr, nullptr, Gr, nullptr,
      MQ, E_, E_, 0, 1.f, nullptr, 0, 0, nullptr, nullptr, nullptr, nullptr);
  gemm_xwt<<<dim3(E_ / 128, MQ / 128), blk, 0, stream>>>(
      out16, E_, 0, wyr16, E_, byr, Gr, Gr, nullptr,
      MQ, E_, E_, 1, 1.f, nullptr, 0, 0, nullptr, nullptr, nullptr, nullptr);
  // z = relu(oriq@wxz^T + bxz + out@wyz^T + byz)
  gemm_xwt<<<dim3(E_ / 128, MQ / 128), blk, 0, stream>>>(
      xv16, E_, 1, wxz16, E_, bxz, nullptr, Gz, nullptr,
      MQ, E_, E_, 0, 1.f, nullptr, 0, 0, nullptr, nullptr, nullptr, nullptr);
  gemm_xwt<<<dim3(E_ / 128, MQ / 128), blk, 0, stream>>>(
      out16, E_, 0, wyz16, E_, byz, Gz, Gz, nullptr,
      MQ, E_, E_, 1, 1.f, nullptr, 0, 0, nullptr, nullptr, nullptr, nullptr);
  // U = oriq @ wxg^T (no bias)
  gemm_xwt<<<dim3(E_ / 128, MQ / 128), blk, 0, stream>>>(
      xv16, E_, 1, wxg16, E_, nullptr, nullptr, Uu, nullptr,
      MQ, E_, E_, 0, 1.f, nullptr, 0, 0, nullptr, nullptr, nullptr, nullptr);
  // final: val = out@wyg^T + bxg; output = (1-z)*oriq + z*tanh(r*U + val + byg)
  gemm_xwt<<<dim3(E_ / 128, MQ / 128), blk, 0, stream>>>(
      out16, E_, 0, wyg16, E_, bxg, nullptr, outp, nullptr,
      MQ, E_, E_, 0, 1.f, key, 1, 0, Gr, Gz, Uu, byg);
  #undef NB
}
